// MultiHeadSelfAttention_58695023067648
// MI455X (gfx1250) — matrix operand order repaired and checked by reading, not compiled, not run on hardware
//
#include <hip/hip_runtime.h>

// ---- problem constants (match reference) ----
#define D_MODEL   2048
#define NUM_HEADS 16
#define D_K       128
#define SEQ       2048
#define BATCH     2
#define O3        (3 * D_MODEL)
#define NKSTEP    (D_MODEL / 32)               // 64 K-steps

typedef __attribute__((ext_vector_type(16))) __bf16 v16bf;
typedef __attribute__((ext_vector_type(8)))  float  v8f;
typedef __attribute__((ext_vector_type(4)))  unsigned u32x4;
typedef __attribute__((ext_vector_type(8)))  int      i32x8;
typedef __attribute__((ext_vector_type(4)))  int      i32x4;

#define ROPE_LOG_COEF (-0.143911568f)          // -2*ln(10000)/128
#define QK_SCALE      (0.08838834764831845f)   // 1/sqrt(128)

// Toolchain hedge: therock-10.0 headers (6-arg tensor builtin) ship this file;
// ROCm 7.2 (5-arg builtin) does not.
#if __has_include(<hip/amd_detail/amd_gfx1250_TDM.h>)
#define TDM_6ARG 1
#else
#define TDM_6ARG 0
#endif

// ---------------------------------------------------------------------
// TDM: issue a 1D/2D tensor_load_to_lds.  Descriptor packing per ISA
// CDNA5 ISA (group0) / §8.4 (group1).
// data_size_code: 0=1B,1=2B,2=4B,3=8B.  dims/strides in data_size units.
// tile_d1 == 0 -> 1D tile.
// ---------------------------------------------------------------------
__device__ __forceinline__ void tdm_load_2d(unsigned lds_addr, const void* gptr,
                                            unsigned data_size_code,
                                            unsigned tensor_d0, unsigned tensor_d1,
                                            unsigned tile_d0, unsigned tile_d1,
                                            unsigned long long stride0) {
    unsigned long long ga = (unsigned long long)gptr;
    u32x4 g0;
    g0[0] = 1u;                                                 // count=1 (valid), user mode
    g0[1] = lds_addr;                                           // bits[63:32]
    g0[2] = (unsigned)(ga & 0xFFFFFFFFu);                       // global_addr lo
    g0[3] = (unsigned)((ga >> 32) & 0x01FFFFFFu) | (2u << 30);  // addr hi | type=2
    i32x8 g1;
    g1[0] = (int)(data_size_code << 16);                        // mask=0, no pad/iterate
    g1[1] = (int)((tensor_d0 & 0xFFFFu) << 16);
    g1[2] = (int)((tensor_d0 >> 16) | ((tensor_d1 & 0xFFFFu) << 16));
    g1[3] = (int)((tensor_d1 >> 16) | (tile_d0 << 16));
    g1[4] = (int)tile_d1;                                       // tile_dim2 = 0
    g1[5] = (int)(stride0 & 0xFFFFFFFFull);
    g1[6] = (int)((stride0 >> 32) & 0xFFFFull);                 // stride1 = 0
    g1[7] = 0;
    i32x4 z4 = {0, 0, 0, 0};
#if TDM_6ARG
    i32x8 z8 = {0, 0, 0, 0, 0, 0, 0, 0};
    __builtin_amdgcn_tensor_load_to_lds(g0, g1, z4, z4, z8, 0);
#else
    __builtin_amdgcn_tensor_load_to_lds(g0, g1, z4, z4, 0);
#endif
}

__device__ __forceinline__ unsigned lds_off_u32(const void* p) {
    return (unsigned)(size_t)p;    // LDS aperture: hw uses addr[31:0]
}

// 8 consecutive floats held raw in VGPRs (register prefetch pipeline)
struct F8 { float4 a, b; };
__device__ __forceinline__ F8 ld8(const float* p) {
    F8 r; r.a = *(const float4*)p; r.b = *(const float4*)(p + 4); return r;
}
__device__ __forceinline__ void st8(__bf16* d, const F8& v) {
    d[0] = (__bf16)v.a.x; d[1] = (__bf16)v.a.y; d[2] = (__bf16)v.a.z; d[3] = (__bf16)v.a.w;
    d[4] = (__bf16)v.b.x; d[5] = (__bf16)v.b.y; d[6] = (__bf16)v.b.z; d[7] = (__bf16)v.b.w;
}

// A-fragment (16x32, 16-bit) from LDS, row-major stride 32.
__device__ __forceinline__ v16bf lds_load_a16x32(const __bf16* lds, int row, int half) {
    v16bf a;
    const __bf16* p0 = lds + row * 32 + 16 * half;
    const __bf16* p1 = p0 + 8;
#pragma unroll
    for (int i = 0; i < 8; i++) a[i] = p0[i];
#pragma unroll
    for (int i = 0; i < 8; i++) a[8 + i] = p1[i];
    return a;
}

// B-fragment (32x16, 16-bit) from LDS laid out [n][k] stride 32.
__device__ __forceinline__ v16bf lds_load_b32x16(const __bf16* lds, int col, int half) {
    v16bf b;
    const __bf16* p = lds + col * 32 + 16 * half;
#pragma unroll
    for (int i = 0; i < 16; i++) b[i] = p[i];
    return b;
}

// =====================================================================
// Kernel 1: qkv = x @ W_qkv^T (+RoPE on q,k; q pre-scaled by 1/sqrt(dk))
// 256 thr (8 waves), 128x128 tile, K-step 32, register-prefetch pipeline.
// =====================================================================
__global__ __launch_bounds__(256)
void qkv_rope_kernel(const float* __restrict__ x,
                     const int*   __restrict__ tpos,
                     const float* __restrict__ Wqkv,
                     __bf16* __restrict__ Qb,
                     __bf16* __restrict__ Kb,
                     __bf16* __restrict__ Vt) {
    __shared__ __bf16 la[128 * 32];   // 8 KB
    __shared__ __bf16 lb[128 * 32];   // 8 KB

    const int tid  = threadIdx.x;
    const int lane = tid & 31;
    const int wave = tid >> 5;
    const int half = lane >> 4;
    const int ln   = lane & 15;

    const int mtile = blockIdx.y;
    const int o0    = blockIdx.x * 128;
    const float* xbase = x + (size_t)mtile * 128 * D_MODEL;

    // per-thread staging slots: rows r0 and r0+64, 8-col chunk c8
    const int r0 = tid >> 2;
    const int c8 = (tid & 3) * 8;
    const size_t rowskip = (size_t)64 * D_MODEL;
    const float* pa = xbase + (size_t)r0 * D_MODEL + c8;
    const float* pb = Wqkv + (size_t)(o0 + r0) * D_MODEL + c8;
    __bf16* da0 = la + r0 * 32 + c8;
    __bf16* da1 = la + (r0 + 64) * 32 + c8;
    __bf16* db0 = lb + r0 * 32 + c8;
    __bf16* db1 = lb + (r0 + 64) * 32 + c8;

    // prologue prefetch (k0 = 0)
    F8 ra0 = ld8(pa),           ra1 = ld8(pa + rowskip);
    F8 rb0 = ld8(pb),           rb1 = ld8(pb + rowskip);

    v8f acc[8];
#pragma unroll
    for (int n = 0; n < 8; n++) acc[n] = (v8f){};

    for (int ks = 0; ks < NKSTEP; ks++) {
        __syncthreads();                              // prev compute done
        st8(da0, ra0); st8(da1, ra1);
        st8(db0, rb0); st8(db1, rb1);
        __syncthreads();                              // tiles visible
        if (ks + 1 < NKSTEP) {                        // prefetch next K-step
            const int kn = (ks + 1) * 32;
            ra0 = ld8(pa + kn); ra1 = ld8(pa + kn + rowskip);
            rb0 = ld8(pb + kn); rb1 = ld8(pb + kn + rowskip);
        }
        v16bf a = lds_load_a16x32(la + wave * 512, ln, half);
#pragma unroll
        for (int n = 0; n < 8; n++) {
            v16bf b = lds_load_b32x16(lb, n * 16 + ln, half);
            acc[n] = __builtin_amdgcn_wmma_f32_16x16x32_bf16(false, a, false, b,
                                                             (short)0, acc[n], false, false);
        }
    }

    // ---- epilogue: RoPE + scatter ----
    const int region = (o0 >> 11);                    // uniform per block
    float posv[8];
#pragma unroll
    for (int i = 0; i < 8; i++)
        posv[i] = (float)tpos[(mtile * 128 + wave * 16 + i + 8 * half) & (SEQ - 1)];

#pragma unroll
    for (int n = 0; n < 8; n++) {
        const int og = o0 + n * 16 + ln;
        const int oc = og & (D_MODEL - 1);
        const int h  = oc >> 7;
        const int d  = oc & (D_K - 1);

        float vals[8];
#pragma unroll
        for (int i = 0; i < 8; i++) vals[i] = acc[n][i];

        if (region < 2) {
            const float inv = __expf(ROPE_LOG_COEF * (float)(d >> 1));
#pragma unroll
            for (int i = 0; i < 8; i++) {
                float xv = vals[i];
                float xp = __shfl_xor(xv, 1, 32);     // pair partner d^1 in lane^1
                float s, c;
                __sincosf(posv[i] * inv, &s, &c);
                vals[i] = (d & 1) ? (xv * c + xp * s) : (xv * c - xp * s);
            }
        }

        const int bq = (mtile * 128) >> 11;
        const size_t bh = (size_t)bq * NUM_HEADS + h;
#pragma unroll
        for (int i = 0; i < 8; i++) {
            int t = (mtile * 128 + wave * 16 + i + 8 * half) & (SEQ - 1);
            if (region == 0) {
                Qb[(bh * SEQ + t) * D_K + d] = (__bf16)(vals[i] * QK_SCALE);
            } else if (region == 1) {
                Kb[(bh * SEQ + t) * D_K + d] = (__bf16)vals[i];
            } else {
                Vt[(bh * D_K + d) * SEQ + t] = (__bf16)vals[i];
            }
        }
    }
}

// =====================================================================
// Kernel 2: causal flash attention. 4 waves = 64 query rows per block.
// K/V tiles staged by the TDM, double-buffered (issue jb+1 before
// waiting on jb; TENSORcnt is in-order per wave).
// =====================================================================
__global__ __launch_bounds__(128)
void attn_kernel(const __bf16* __restrict__ Qb,
                 const __bf16* __restrict__ Kb,
                 const __bf16* __restrict__ Vt,
                 __bf16* __restrict__ Ab) {
    __shared__ __bf16 lk[2][32 * 128];   // K tiles [key][d],  2 x 8 KB
    __shared__ __bf16 lv[2][128 * 32];   // V tiles [d][key],  2 x 8 KB
    __shared__ __bf16 lp[4 * 512];       // P tiles, 16x32 per wave

    const int lane = threadIdx.x & 31;
    const int wave = threadIdx.x >> 5;
    const int half = lane >> 4;
    const int ln   = lane & 15;

    const int q0b = blockIdx.x * 64;
    const int q0w = q0b + wave * 16;
    const int h   = blockIdx.y;
    const int b   = blockIdx.z;
    const size_t bh = (size_t)b * NUM_HEADS + h;

    // resident Q A-fragments
    v16bf qf[4];
    {
        const __bf16* qrow = Qb + (bh * SEQ + q0w + ln) * D_K;
#pragma unroll
        for (int kk = 0; kk < 4; kk++) {
            const __bf16* p0 = qrow + kk * 32 + 16 * half;
            const __bf16* p1 = p0 + 8;
#pragma unroll
            for (int i = 0; i < 8; i++) qf[kk][i] = p0[i];
#pragma unroll
            for (int i = 0; i < 8; i++) qf[kk][8 + i] = p1[i];
        }
    }

    v8f oacc[8];
#pragma unroll
    for (int dt = 0; dt < 8; dt++) oacc[dt] = (v8f){};
    float mrow[8], lrow[8];
#pragma unroll
    for (int i = 0; i < 8; i++) { mrow[i] = -3.0e38f; lrow[i] = 0.0f; }

    unsigned lk_off[2] = { lds_off_u32(lk[0]), lds_off_u32(lk[1]) };
    unsigned lv_off[2] = { lds_off_u32(lv[0]), lds_off_u32(lv[1]) };
    __bf16* lpw = lp + wave * 512;

    const int nkb = (q0b + 95) >> 5;

    // prologue: issue tile 0
    if (wave == 0) {
        tdm_load_2d(lk_off[0], Kb + bh * SEQ * D_K, 3u, 1u << 20, 0u, 1024u, 0u, 0ull);
        tdm_load_2d(lv_off[0], Vt + bh * D_K * SEQ, 3u, 512u, 128u, 8u, 128u, 512ull);
    }

    for (int jb = 0; jb < nkb; jb++) {
        const int j0  = jb * 32;
        const int cur = jb & 1;

        if (jb + 1 < nkb) {
            if (wave == 0) {
                const int jn = j0 + 32;
                tdm_load_2d(lk_off[1 - cur], Kb + (bh * SEQ + jn) * D_K,
                            3u, 1u << 20, 0u, 1024u, 0u, 0ull);
                tdm_load_2d(lv_off[1 - cur], Vt + bh * D_K * SEQ + jn,
                            3u, 512u, 128u, 8u, 128u, 512ull);
                __builtin_amdgcn_s_wait_tensorcnt(2);   // current pair complete
            }
        } else if (wave == 0) {
            __builtin_amdgcn_s_wait_tensorcnt(0);
        }
        __syncthreads();                                 // tiles visible

        if (j0 <= q0w + 15) {                            // wave-uniform causal guard
            const __bf16* lkc = lk[cur];
            const __bf16* lvc = lv[cur];

            // ---- S = Q K^T ----
            v8f sa = {}, sb = {};
#pragma unroll
            for (int kk = 0; kk < 4; kk++) {
                v16bf b0, b1;
                const __bf16* p0 = lkc + (size_t)ln * 128 + kk * 32 + 16 * half;
                const __bf16* p1 = lkc + (size_t)(16 + ln) * 128 + kk * 32 + 16 * half;
#pragma unroll
                for (int i = 0; i < 16; i++) b0[i] = p0[i];
#pragma unroll
                for (int i = 0; i < 16; i++) b1[i] = p1[i];
                sa = __builtin_amdgcn_wmma_f32_16x16x32_bf16(false, qf[kk], false, b0,
                                                             (short)0, sa, false, false);
                sb = __builtin_amdgcn_wmma_f32_16x16x32_bf16(false, qf[kk], false, b1,
                                                             (short)0, sb, false, false);
            }

            // ---- causal mask + online softmax ----
            const int keyA = j0 + ln;
            const int keyB = j0 + 16 + ln;
#pragma unroll
            for (int i = 0; i < 8; i++) {
                const int qr = q0w + i + 8 * half;
                float va = (keyA <= qr) ? sa[i] : -3.0e38f;
                float vb = (keyB <= qr) ? sb[i] : -3.0e38f;

                float mx = fmaxf(va, vb);
#pragma unroll
                for (int m = 1; m < 16; m <<= 1) mx = fmaxf(mx, __shfl_xor(mx, m, 32));
                float mnew = fmaxf(mrow[i], mx);
                float corr = __expf(mrow[i] - mnew);
                float pa = __expf(va - mnew);
                float pb = __expf(vb - mnew);
                float rs = pa + pb;
#pragma unroll
                for (int m = 1; m < 16; m <<= 1) rs += __shfl_xor(rs, m, 32);
                lrow[i] = lrow[i] * corr + rs;
                mrow[i] = mnew;
#pragma unroll
                for (int dt = 0; dt < 8; dt++) oacc[dt][i] *= corr;

                lpw[(i + 8 * half) * 32 + ln]      = (__bf16)pa;
                lpw[(i + 8 * half) * 32 + 16 + ln] = (__bf16)pb;
            }

            // ---- O += P * V ----
            v16bf pf = lds_load_a16x32(lpw, ln, half);
#pragma unroll
            for (int dt = 0; dt < 8; dt++) {
                v16bf vf;
                const __bf16* vp = lvc + (size_t)(dt * 16 + ln) * 32 + 16 * half;
#pragma unroll
                for (int i = 0; i < 16; i++) vf[i] = vp[i];
                oacc[dt] = __builtin_amdgcn_wmma_f32_16x16x32_bf16(false, pf, false, vf,
                                                                   (short)0, oacc[dt], false, false);
            }
        }
        __syncthreads();           // all readers done before buffer is re-filled
    }

    // ---- normalize + write concat buffer ----
#pragma unroll
    for (int dt = 0; dt < 8; dt++) {
#pragma unroll
        for (int i = 0; i < 8; i++) {
            float v = oacc[dt][i] / lrow[i];
            int t = q0w + i + 8 * half;
            Ab[((size_t)b * SEQ + t) * D_MODEL + h * D_K + dt * 16 + ln] = (__bf16)v;
        }
    }
}

// =====================================================================
// Kernel 3: out = concat @ W_out^T.  TDM-staged A tile (double-buffered)
// + register-prefetch pipeline for the fp32 W tile.
// =====================================================================
__global__ __launch_bounds__(256)
void out_proj_kernel(const __bf16* __restrict__ Ab,
                     const float*  __restrict__ Wout,
                     float* __restrict__ out) {
    __shared__ __bf16 la2[2][128 * 32];   // 2 x 8 KB (TDM ping-pong)
    __shared__ __bf16 lb[128 * 32];

    const int tid  = threadIdx.x;
    const int lane = tid & 31;
    const int wave = tid >> 5;
    const int half = lane >> 4;
    const int ln   = lane & 15;

    const int mtile = blockIdx.y;
    const int o0    = blockIdx.x * 128;
    const __bf16* abase = Ab + (size_t)mtile * 128 * D_MODEL;
    unsigned la_off[2] = { lds_off_u32(la2[0]), lds_off_u32(la2[1]) };

    const int r0 = tid >> 2;
    const int c8 = (tid & 3) * 8;
    const size_t rowskip = (size_t)64 * D_MODEL;
    const float* pb = Wout + (size_t)(o0 + r0) * D_MODEL + c8;
    __bf16* db0 = lb + r0 * 32 + c8;
    __bf16* db1 = lb + (r0 + 64) * 32 + c8;

    F8 rb0 = ld8(pb), rb1 = ld8(pb + rowskip);
    if (wave == 0)        // prologue: A tile for k0=0
        tdm_load_2d(la_off[0], abase, 3u, 512u, 128u, 8u, 128u, 512ull);

    v8f acc[8];
#pragma unroll
    for (int n = 0; n < 8; n++) acc[n] = (v8f){};

    for (int ks = 0; ks < NKSTEP; ks++) {
        const int cur = ks & 1;
        __syncthreads();                              // prev compute done
        st8(db0, rb0); st8(db1, rb1);
        if (ks + 1 < NKSTEP) {
            if (wave == 0) {
                tdm_load_2d(la_off[1 - cur], abase + (ks + 1) * 32,
                            3u, 512u, 128u, 8u, 128u, 512ull);
                __builtin_amdgcn_s_wait_tensorcnt(1);  // current A tile complete
            }
        } else if (wave == 0) {
            __builtin_amdgcn_s_wait_tensorcnt(0);
        }
        __syncthreads();                              // lb + la2[cur] visible
        if (ks + 1 < NKSTEP) {                        // prefetch next W chunk
            const int kn = (ks + 1) * 32;
            rb0 = ld8(pb + kn); rb1 = ld8(pb + kn + rowskip);
        }

        v16bf a = lds_load_a16x32(la2[cur] + wave * 512, ln, half);
#pragma unroll
        for (int n = 0; n < 8; n++) {
            v16bf b = lds_load_b32x16(lb, n * 16 + ln, half);
            acc[n] = __builtin_amdgcn_wmma_f32_16x16x32_bf16(false, a, false, b,
                                                             (short)0, acc[n], false, false);
        }
    }

#pragma unroll
    for (int n = 0; n < 8; n++) {
#pragma unroll
        for (int i = 0; i < 8; i++) {
            int gr = mtile * 128 + wave * 16 + i + 8 * half;
            out[(size_t)gr * D_MODEL + o0 + n * 16 + ln] = acc[n][i];
        }
    }
}

// =====================================================================
// Host launcher
// =====================================================================
extern "C" void kernel_launch(void* const* d_in, const int* in_sizes, int n_in,
                              void* d_out, int out_size, void* d_ws, size_t ws_size,
                              hipStream_t stream) {
    (void)in_sizes; (void)n_in; (void)out_size; (void)ws_size;

    const float* x    = (const float*)d_in[0];
    const int*   tpos = (const int*)  d_in[1];
    const float* Wqkv = (const float*)d_in[2];
    const float* Wout = (const float*)d_in[3];
    float*       out  = (float*)d_out;

    const size_t QK_ELEMS = (size_t)BATCH * NUM_HEADS * SEQ * D_K;   // 8M elems
    __bf16* Qb = (__bf16*)d_ws;
    __bf16* Kb = Qb + QK_ELEMS;
    __bf16* Vt = Kb + QK_ELEMS;
    __bf16* Ab = Vt + QK_ELEMS;

    {   // 1) QKV projection + RoPE
        dim3 grid(O3 / 128, (BATCH * SEQ) / 128);       // (48, 32)
        qkv_rope_kernel<<<grid, dim3(256), 0, stream>>>(x, tpos, Wqkv, Qb, Kb, Vt);
    }
    {   // 2) causal attention
        dim3 grid(SEQ / 64, NUM_HEADS, BATCH);          // (32, 16, 2)
        attn_kernel<<<grid, dim3(128), 0, stream>>>(Qb, Kb, Vt, Ab);
    }
    {   // 3) output projection
        dim3 grid(D_MODEL / 128, (BATCH * SEQ) / 128);  // (16, 32)
        out_proj_kernel<<<grid, dim3(256), 0, stream>>>(Ab, Wout, out);
    }
}
